// TransformerBlock_7181185319069
// MI455X (gfx1250) — matrix operand order repaired and checked by reading, not compiled, not run on hardware
//
#include <hip/hip_runtime.h>
#include <hip/hip_bf16.h>

typedef __bf16 bf16;
typedef __attribute__((ext_vector_type(8)))  bf16  bf16x8;
typedef __attribute__((ext_vector_type(16))) bf16  v16bf;
typedef __attribute__((ext_vector_type(8)))  float v8f;
typedef __attribute__((ext_vector_type(4)))  int   i32x4;

#define DMODEL 768
#define NHEADS 12
#define DHEAD  64
#define DMLP   3072
#define BATCHN 2
#define SEQLEN 2048
#define ROWS   (BATCHN * SEQLEN)   // 4096

// ---- CDNA5 async global->LDS copy (guarded; falls back to reg bounce) ----
#if defined(__has_builtin)
#if __has_builtin(__builtin_amdgcn_global_load_async_to_lds_b128) && \
    __has_builtin(__builtin_amdgcn_s_wait_asynccnt)
#define ASYNC_LDS 1
#endif
#endif

static __device__ __forceinline__ void copy16_g2l(const bf16* g, bf16* l) {
#ifdef ASYNC_LDS
  __builtin_amdgcn_global_load_async_to_lds_b128(
      (__attribute__((address_space(1))) i32x4*)g,
      (__attribute__((address_space(3))) i32x4*)l, 0, 0);
#else
  *(bf16x8*)l = *(const bf16x8*)g;
#endif
}

static __device__ __forceinline__ void wait_async() {
#ifdef ASYNC_LDS
  __builtin_amdgcn_s_wait_asynccnt(0);
#endif
}

static __device__ __forceinline__ v16bf concat8(bf16x8 lo, bf16x8 hi) {
  return __builtin_shufflevector(lo, hi, 0,1,2,3,4,5,6,7,8,9,10,11,12,13,14,15);
}

static __device__ __forceinline__ float gelu_new(float x) {
  const float c = 0.7978845608028654f; // sqrt(2/pi)
  float x3 = x * x * x;
  return 0.5f * x * (1.0f + tanhf(c * (x + 0.044715f * x3)));
}

// ---------------------------------------------------------------------------
// Weight conversion kernels
// ---------------------------------------------------------------------------
__global__ void cast_f32_bf16(const float* __restrict__ in, bf16* __restrict__ out, int n) {
  for (int i = blockIdx.x * blockDim.x + threadIdx.x; i < n; i += gridDim.x * blockDim.x)
    out[i] = (bf16)in[i];
}

// W[h, d, e] (12,768,64) -> out[d, h*64+e] as a row-major [768,768] B matrix
__global__ void convert_qkv_w(const float* __restrict__ in, bf16* __restrict__ out, int n) {
  for (int i = blockIdx.x * blockDim.x + threadIdx.x; i < n; i += gridDim.x * blockDim.x) {
    int h = i / (DMODEL * DHEAD);
    int d = (i / DHEAD) % DMODEL;
    int e = i % DHEAD;
    out[(size_t)d * (NHEADS * DHEAD) + h * DHEAD + e] = (bf16)in[i];
  }
}

// ---------------------------------------------------------------------------
// LayerNorm: one 256-thread block per row of 768, bf16 output
// ---------------------------------------------------------------------------
__global__ __launch_bounds__(256)
void layernorm_kernel(const float* __restrict__ in, const float* __restrict__ w,
                      const float* __restrict__ b, bf16* __restrict__ out) {
  const int row = blockIdx.x;
  const float* x = in + (size_t)row * DMODEL;
  const int tid = threadIdx.x, lane = tid & 31, wv = tid >> 5;
  __shared__ float red1[8], red2[8];

  float s = 0.f;
  for (int i = tid; i < DMODEL; i += 256) s += x[i];
  for (int off = 16; off; off >>= 1) s += __shfl_xor(s, off, 32);
  if (lane == 0) red1[wv] = s;
  __syncthreads();
  float mean = 0.f;
  #pragma unroll
  for (int j = 0; j < 8; ++j) mean += red1[j];
  mean *= (1.0f / DMODEL);

  float v = 0.f;
  for (int i = tid; i < DMODEL; i += 256) { float d = x[i] - mean; v += d * d; }
  for (int off = 16; off; off >>= 1) v += __shfl_xor(v, off, 32);
  if (lane == 0) red2[wv] = v;
  __syncthreads();
  float var = 0.f;
  #pragma unroll
  for (int j = 0; j < 8; ++j) var += red2[j];
  var *= (1.0f / DMODEL);
  float sc = rsqrtf(var + 1e-5f);

  bf16* o = out + (size_t)row * DMODEL;
  for (int i = tid; i < DMODEL; i += 256)
    o[i] = (bf16)(((x[i] - mean) * sc) * w[i] + b[i]);
}

// ---------------------------------------------------------------------------
// bf16 WMMA GEMM, double-buffered LDS + async A copies.
// Block tile 256x64 (8 waves), wave tile 32x64 -> 8 WMMAs per 32-k step.
// ---------------------------------------------------------------------------
template<bool GELU, bool HAS_RESID, bool OUT_BF16>
__global__ __launch_bounds__(256)
void gemm_bf16(const bf16* __restrict__ A, const bf16* __restrict__ B,
               const float* __restrict__ bias, const float* __restrict__ resid,
               float* __restrict__ outF, bf16* __restrict__ outB,
               int M, int N, int K) {
  __shared__ __align__(16) bf16 Asm[2][256][40];   // [buf][m][k], 80B rows
  __shared__ __align__(16) bf16 BsT[2][64][40];    // [buf][n][k] (transposed)

  const int tid  = threadIdx.x;
  const int lane = tid & 31;
  const int wv   = tid >> 5;
  const int hl   = lane >> 4;      // lane-half
  const int r    = lane & 15;
  const int m0   = blockIdx.y * 256;
  const int n0   = blockIdx.x * 64;

  auto loadA = [&](int k0, int buf) {
    #pragma unroll
    for (int it0 = 0; it0 < 4; ++it0) {            // 256x32 elems / 8 / 256thr
      int it  = it0 * 256 + tid;
      int row = it >> 2;
      int kk  = (it & 3) * 8;
      copy16_g2l(A + (size_t)(m0 + row) * K + k0 + kk, &Asm[buf][row][kk]);
    }
  };
  auto loadB = [&](int k0, int buf) {
    int kk = tid >> 3;
    int nn = (tid & 7) * 8;
    bf16x8 vv = *(const bf16x8*)(B + (size_t)(k0 + kk) * N + n0 + nn);
    #pragma unroll
    for (int j = 0; j < 8; ++j) BsT[buf][nn + j][kk] = vv[j];
  };

  v8f acc[2][4];
  #pragma unroll
  for (int m = 0; m < 2; ++m)
    #pragma unroll
    for (int c = 0; c < 4; ++c)
      #pragma unroll
      for (int j = 0; j < 8; ++j) acc[m][c][j] = 0.f;

  loadA(0, 0);
  loadB(0, 0);

  const int nsteps = K / 32;
  for (int s = 0; s < nsteps; ++s) {
    const int buf = s & 1;
    wait_async();        // my tile-s async transfers have landed in LDS
    __syncthreads();     // everyone's tile-s landed; buf^1 free to overwrite
    if (s + 1 < nsteps) { loadA((s + 1) * 32, buf ^ 1); loadB((s + 1) * 32, buf ^ 1); }

    // A fragments (ISA 16-bit 16x32 layout): row = r; k = (i<8?i:i+8)+8*hl
    const bf16* ap0 = &Asm[buf][wv * 32 + r][0];
    const bf16* ap1 = &Asm[buf][wv * 32 + 16 + r][0];
    v16bf a0 = concat8(*(const bf16x8*)(ap0 + 8 * hl),
                       *(const bf16x8*)(ap0 + 16 + 8 * hl));
    v16bf a1 = concat8(*(const bf16x8*)(ap1 + 8 * hl),
                       *(const bf16x8*)(ap1 + 16 + 8 * hl));
    #pragma unroll
    for (int c = 0; c < 4; ++c) {
      // B fragment: col = r; k = (i<8?i:i+8)+8*hl (same order as A)
      const bf16* bp = &BsT[buf][c * 16 + r][0];
      v16bf bfrag = concat8(*(const bf16x8*)(bp + 8 * hl),
                            *(const bf16x8*)(bp + 16 + 8 * hl));
      acc[0][c] = __builtin_amdgcn_wmma_f32_16x16x32_bf16(
          false, a0, false, bfrag, (short)0, acc[0][c], false, false);
      acc[1][c] = __builtin_amdgcn_wmma_f32_16x16x32_bf16(
          false, a1, false, bfrag, (short)0, acc[1][c], false, false);
    }
  }

  // Epilogue: acc[m][c][v] = (row = m0+wv*32+16m+v+8*hl, col = n0+c*16+r)
  #pragma unroll
  for (int m = 0; m < 2; ++m) {
    #pragma unroll
    for (int c = 0; c < 4; ++c) {
      #pragma unroll
      for (int v = 0; v < 8; ++v) {
        int row = m0 + wv * 32 + 16 * m + v + 8 * hl;
        int col = n0 + c * 16 + r;
        float val = acc[m][c][v] + bias[col];
        if (GELU) val = gelu_new(val);
        if (HAS_RESID) val += resid[(size_t)row * N + col];
        if (OUT_BF16) outB[(size_t)row * N + col] = (bf16)val;
        else          outF[(size_t)row * N + col] = val;
      }
    }
  }
}

// ---------------------------------------------------------------------------
// Flash attention: block = 4 waves = 4 consecutive 16-row q tiles of one (b,h)
// K/V staged in LDS per 32-key block (K via async copy); online softmax.
// ---------------------------------------------------------------------------
__global__ __launch_bounds__(128)
void attn_kernel(const bf16* __restrict__ q, const bf16* __restrict__ k,
                 const bf16* __restrict__ v, bf16* __restrict__ z) {
  __shared__ __align__(16) bf16 Ksm[32][72];    // [key][e]
  __shared__ __align__(16) bf16 VsmT[64][40];   // [e][key]
  __shared__ __align__(16) bf16 Psm[4][16][40]; // per-wave P re-layout

  const int bid  = blockIdx.x;
  const int qblk = bid % (SEQLEN / 64);
  const int h    = (bid / (SEQLEN / 64)) % NHEADS;
  const int bb   = bid / ((SEQLEN / 64) * NHEADS);
  const int tid  = threadIdx.x;
  const int lane = tid & 31;
  const int wv   = tid >> 5;
  const int hl   = lane >> 4;
  const int r    = lane & 15;

  const int m0b = qblk * 64;
  const int m0w = m0b + wv * 16;
  const size_t seqbase = (size_t)bb * SEQLEN;

  // Q fragments: A 16x32 layout over e chunks [0,32) and [32,64)
  const bf16* qrow = q + (seqbase + m0w + r) * DMODEL + h * DHEAD;
  v16bf aq0 = concat8(*(const bf16x8*)(qrow + 8 * hl),
                      *(const bf16x8*)(qrow + 16 + 8 * hl));
  v16bf aq1 = concat8(*(const bf16x8*)(qrow + 32 + 8 * hl),
                      *(const bf16x8*)(qrow + 48 + 8 * hl));

  v8f zacc[4];
  float Mst[8], Lst[8];
  #pragma unroll
  for (int c = 0; c < 4; ++c)
    #pragma unroll
    for (int j = 0; j < 8; ++j) zacc[c][j] = 0.f;
  #pragma unroll
  for (int j = 0; j < 8; ++j) { Mst[j] = -3.0e38f; Lst[j] = 0.f; }

  const int nkb = qblk * 2 + 2;
  for (int kb = 0; kb < nkb; ++kb) {
    const int kbase = kb * 32;
    __syncthreads();                    // previous-iteration readers done
    for (int it = tid; it < 256; it += 128) {
      int kk = it >> 3;
      int ee = (it & 7) * 8;
      size_t g = (seqbase + kbase + kk) * DMODEL + h * DHEAD + ee;
      copy16_g2l(k + g, &Ksm[kk][ee]);            // async K tile
      bf16x8 vv = *(const bf16x8*)(v + g);        // V transposed manually
      #pragma unroll
      for (int j = 0; j < 8; ++j) VsmT[ee + j][kk] = vv[j];
    }
    wait_async();
    __syncthreads();

    if (kbase <= m0w + 15) {            // wave-uniform: EXEC stays all-1
      v8f sc[2];
      #pragma unroll
      for (int g = 0; g < 2; ++g) {
        #pragma unroll
        for (int j = 0; j < 8; ++j) sc[g][j] = 0.f;
        const bf16* kp = &Ksm[g * 16 + r][0];
        v16bf bk0 = concat8(*(const bf16x8*)(kp + 8 * hl),
                            *(const bf16x8*)(kp + 16 + 8 * hl));
        v16bf bk1 = concat8(*(const bf16x8*)(kp + 32 + 8 * hl),
                            *(const bf16x8*)(kp + 48 + 8 * hl));
        sc[g] = __builtin_amdgcn_wmma_f32_16x16x32_bf16(
            false, aq0, false, bk0, (short)0, sc[g], false, false);
        sc[g] = __builtin_amdgcn_wmma_f32_16x16x32_bf16(
            false, aq1, false, bk1, (short)0, sc[g], false, false);
      }
      // online softmax per q row (rows live on one lane-half; reduce over 16 lanes)
      const int key0 = kbase + r, key1 = kbase + 16 + r;
      #pragma unroll
      for (int vv_ = 0; vv_ < 8; ++vv_) {
        int qr = m0w + vv_ + 8 * hl;
        float s0 = (key0 <= qr) ? sc[0][vv_] * 0.125f : -1.0e30f;
        float s1 = (key1 <= qr) ? sc[1][vv_] * 0.125f : -1.0e30f;
        float mx = fmaxf(s0, s1);
        #pragma unroll
        for (int off = 1; off <= 8; off <<= 1) mx = fmaxf(mx, __shfl_xor(mx, off, 32));
        float Mnew = fmaxf(Mst[vv_], mx);
        float corr = __expf(Mst[vv_] - Mnew);
        float p0 = __expf(s0 - Mnew);
        float p1 = __expf(s1 - Mnew);
        float rs = p0 + p1;
        #pragma unroll
        for (int off = 1; off <= 8; off <<= 1) rs += __shfl_xor(rs, off, 32);
        Lst[vv_] = Lst[vv_] * corr + rs;
        Mst[vv_] = Mnew;
        #pragma unroll
        for (int c = 0; c < 4; ++c) zacc[c][vv_] *= corr;
        Psm[wv][vv_ + 8 * hl][r]      = (bf16)p0;   // C-layout -> LDS
        Psm[wv][vv_ + 8 * hl][16 + r] = (bf16)p1;
      }
      // P back as A fragment (16q x 32key); wave-private LDS, in-order DS
      const bf16* pp = &Psm[wv][r][0];
      v16bf ap = concat8(*(const bf16x8*)(pp + 8 * hl),
                         *(const bf16x8*)(pp + 16 + 8 * hl));
      #pragma unroll
      for (int c = 0; c < 4; ++c) {
        const bf16* vp = &VsmT[c * 16 + r][0];
        v16bf bv = concat8(*(const bf16x8*)(vp + 8 * hl),
                           *(const bf16x8*)(vp + 16 + 8 * hl));
        zacc[c] = __builtin_amdgcn_wmma_f32_16x16x32_bf16(
            false, ap, false, bv, (short)0, zacc[c], false, false);
      }
    }
  }

  #pragma unroll
  for (int vv_ = 0; vv_ < 8; ++vv_) {
    float inv = 1.0f / Lst[vv_];
    size_t row = seqbase + m0w + vv_ + 8 * hl;
    #pragma unroll
    for (int c = 0; c < 4; ++c)
      z[row * DMODEL + h * DHEAD + c * 16 + r] = (bf16)(zacc[c][vv_] * inv);
  }
}

// ---------------------------------------------------------------------------
// Host launcher
// ---------------------------------------------------------------------------
extern "C" void kernel_launch(void* const* d_in, const int* in_sizes, int n_in,
                              void* d_out, int out_size, void* d_ws, size_t ws_size,
                              hipStream_t stream) {
  const float* resid_pre = (const float*)d_in[0];
  const float* W_Q  = (const float*)d_in[1];
  const float* b_Q  = (const float*)d_in[2];
  const float* W_K  = (const float*)d_in[3];
  const float* b_K  = (const float*)d_in[4];
  const float* W_V  = (const float*)d_in[5];
  const float* b_V  = (const float*)d_in[6];
  const float* W_O  = (const float*)d_in[7];
  const float* b_O  = (const float*)d_in[8];
  const float* ln1w = (const float*)d_in[9];
  const float* ln1b = (const float*)d_in[10];
  const float* ln2w = (const float*)d_in[11];
  const float* ln2b = (const float*)d_in[12];
  const float* W_in = (const float*)d_in[13];
  const float* b_in = (const float*)d_in[14];
  const float* W_out= (const float*)d_in[15];
  const float* b_out= (const float*)d_in[16];
  float* out = (float*)d_out;

  char* wsb = (char*)d_ws;
  size_t off = 0;
  auto take = [&](size_t bytes) -> void* {
    void* p = wsb + off;
    off += (bytes + 255) & ~(size_t)255;
    return p;
  };
  const size_t nWqkv = (size_t)NHEADS * DMODEL * DHEAD;  // 589824
  const size_t nWmlp = (size_t)DMODEL * DMLP;            // 2359296
  const size_t nAct  = (size_t)ROWS * DMODEL;            // 3145728
  const size_t nHid  = (size_t)ROWS * DMLP;              // 12582912

  bf16* WQb   = (bf16*)take(nWqkv * 2);
  bf16* WKb   = (bf16*)take(nWqkv * 2);
  bf16* WVb   = (bf16*)take(nWqkv * 2);
  bf16* WOb   = (bf16*)take(nWqkv * 2);
  bf16* Winb  = (bf16*)take(nWmlp * 2);
  bf16* Woutb = (bf16*)take(nWmlp * 2);
  bf16* xb    = (bf16*)take(nAct * 2);
  bf16* qb    = (bf16*)take(nAct * 2);
  bf16* kbuf  = (bf16*)take(nAct * 2);
  bf16* vbuf  = (bf16*)take(nAct * 2);
  bf16* zb    = (bf16*)take(nAct * 2);
  bf16* yb    = (bf16*)take(nAct * 2);
  float* rm   = (float*)take(nAct * 4);
  bf16* hb    = (bf16*)take(nHid * 2);

  // 1) weight conversion
  convert_qkv_w<<<512, 256, 0, stream>>>(W_Q, WQb, (int)nWqkv);
  convert_qkv_w<<<512, 256, 0, stream>>>(W_K, WKb, (int)nWqkv);
  convert_qkv_w<<<512, 256, 0, stream>>>(W_V, WVb, (int)nWqkv);
  cast_f32_bf16<<<512, 256, 0, stream>>>(W_O,  WOb,  (int)nWqkv);
  cast_f32_bf16<<<512, 256, 0, stream>>>(W_in, Winb, (int)nWmlp);
  cast_f32_bf16<<<512, 256, 0, stream>>>(W_out,Woutb,(int)nWmlp);

  // 2) LN1
  layernorm_kernel<<<ROWS, 256, 0, stream>>>(resid_pre, ln1w, ln1b, xb);

  // 3) QKV projections
  dim3 g768(DMODEL / 64, ROWS / 256);
  gemm_bf16<false, false, true><<<g768, 256, 0, stream>>>(
      xb, WQb, b_Q, nullptr, nullptr, qb, ROWS, DMODEL, DMODEL);
  gemm_bf16<false, false, true><<<g768, 256, 0, stream>>>(
      xb, WKb, b_K, nullptr, nullptr, kbuf, ROWS, DMODEL, DMODEL);
  gemm_bf16<false, false, true><<<g768, 256, 0, stream>>>(
      xb, WVb, b_V, nullptr, nullptr, vbuf, ROWS, DMODEL, DMODEL);

  // 4) attention
  attn_kernel<<<BATCHN * NHEADS * (SEQLEN / 64), 128, 0, stream>>>(qb, kbuf, vbuf, zb);

  // 5) W_O + residual -> resid_mid (f32)
  gemm_bf16<false, true, false><<<g768, 256, 0, stream>>>(
      zb, WOb, b_O, resid_pre, rm, nullptr, ROWS, DMODEL, DMODEL);

  // 6) LN2
  layernorm_kernel<<<ROWS, 256, 0, stream>>>(rm, ln2w, ln2b, yb);

  // 7) MLP in + GELU -> bf16
  dim3 g3072(DMLP / 64, ROWS / 256);
  gemm_bf16<true, false, true><<<g3072, 256, 0, stream>>>(
      yb, Winb, b_in, nullptr, nullptr, hb, ROWS, DMLP, DMODEL);

  // 8) MLP out + resid_mid -> d_out (f32)
  gemm_bf16<false, true, false><<<g768, 256, 0, stream>>>(
      hb, Woutb, b_out, rm, out, nullptr, ROWS, DMODEL, DMLP);
}
